// SourceDecoder_10565619548406
// MI455X (gfx1250) — matrix operand order repaired and checked by reading, not compiled, not run on hardware
//
#include <hip/hip_runtime.h>
#include <hip/hip_bf16.h>

// ---------------------------------------------------------------------------
// Types for CDNA5 WMMA
// ---------------------------------------------------------------------------
typedef __attribute__((ext_vector_type(16))) _Float16 v16h;
typedef __attribute__((ext_vector_type(8)))  _Float16 h8;
typedef __attribute__((ext_vector_type(4)))  _Float16 h4;
typedef __attribute__((ext_vector_type(8)))  float    v8f;
typedef __attribute__((ext_vector_type(4)))  float    f4;

#define BM 128
#define BN 128
#define BK 32
#define LDS_STR 48   // halves per LDS row (32 data + pad), 96 B: keeps 16/32B alignment

// ---------------------------------------------------------------------------
// Generic GEMM: C[M,N] = act(A[M,K] @ W[N,K]^T + bias) (+ resid)
// A, W fp32 in global; f16 tiles in double-buffered LDS; f32 WMMA accumulate.
// 256 threads = 8 waves; 8x8 grid of 16x16 subtiles; each wave owns 4x2.
// FULLN: N is a multiple of BN -> unconditional staging loads / stores.
// ---------------------------------------------------------------------------
template <bool FULLN>
__global__ __launch_bounds__(256) void gemm_wmma(
    const float* __restrict__ A, const float* __restrict__ W,
    const float* __restrict__ bias, const float* __restrict__ resid,
    float* __restrict__ C, int M, int N, int K, int gelu)
{
    __shared__ _Float16 Ash[2][BM * LDS_STR];   // [m][k]
    __shared__ _Float16 Bsh[2][BN * LDS_STR];   // [n][k]  (n-major)

    const int t     = threadIdx.x;
    const int lane  = t & 31;
    const int wave  = t >> 5;
    const int tileM = blockIdx.y * BM;
    const int tileN = blockIdx.x * BN;

    const int wrow0 = (wave >> 2) * 4;   // subtile rows wrow0..wrow0+3
    const int wcol0 = (wave & 3) * 2;    // subtile cols wcol0..wcol0+1
    const int l15 = lane & 31 & 15;
    const int hi  = lane >> 4;

    // Per-thread staging coordinates: 4 float4 quads for A and for B.
    int rowq[4], kqq[4];
    #pragma unroll
    for (int i = 0; i < 4; ++i) {
        int q = i * 256 + t;
        rowq[i] = q >> 3;           // 8 quads per 32-wide row
        kqq[i]  = (q & 7) * 4;
    }

    v8f acc[4][2];
    #pragma unroll
    for (int rr = 0; rr < 4; ++rr)
        #pragma unroll
        for (int cc = 0; cc < 2; ++cc) acc[rr][cc] = (v8f){};

    f4 av[4], bv[4];
    auto stage = [&](int k0) {
        #pragma unroll
        for (int i = 0; i < 4; ++i)
            av[i] = *(const f4*)(A + (size_t)(tileM + rowq[i]) * K + k0 + kqq[i]);
        #pragma unroll
        for (int i = 0; i < 4; ++i) {
            f4 v = (f4){};
            if (FULLN || (tileN + rowq[i] < N))
                v = *(const f4*)(W + (size_t)(tileN + rowq[i]) * K + k0 + kqq[i]);
            bv[i] = v;
        }
    };
    auto commit = [&](int buf) {
        #pragma unroll
        for (int i = 0; i < 4; ++i)
            *(h4*)&Ash[buf][rowq[i] * LDS_STR + kqq[i]] =
                __builtin_convertvector(av[i], h4);
        #pragma unroll
        for (int i = 0; i < 4; ++i)
            *(h4*)&Bsh[buf][rowq[i] * LDS_STR + kqq[i]] =
                __builtin_convertvector(bv[i], h4);
    };

    stage(0);
    commit(0);
    __syncthreads();

    const int nsteps = K / BK;
    for (int st = 0; st < nsteps; ++st) {
        const int cur = st & 1;
        // Issue next tile's global loads before computing on the current one.
        if (st + 1 < nsteps) stage((st + 1) * BK);

        // ---- Build fragments (contiguous b128 LDS loads) ----
        // A 16x32 layout (ISA 7.12.2): K(e,hi) = (e>>3)*16 + hi*8 + (e&7)
        v16h af[4];
        #pragma unroll
        for (int rr = 0; rr < 4; ++rr) {
            int am = 16 * (wrow0 + rr) + l15;
            h8 lo = *(const h8*)&Ash[cur][am * LDS_STR + hi * 8];
            h8 hh = *(const h8*)&Ash[cur][am * LDS_STR + hi * 8 + 16];
            af[rr] = __builtin_shufflevector(lo, hh, 0, 1, 2, 3, 4, 5, 6, 7,
                                             8, 9, 10, 11, 12, 13, 14, 15);
        }
        // B 32x16 layout: lane n = l15, K(e,hi) = (e>>3)*16 + hi*8 + (e&7)
        v16h bf[2];
        #pragma unroll
        for (int cc = 0; cc < 2; ++cc) {
            int bn = 16 * (wcol0 + cc) + l15;
            h8 blo = *(const h8*)&Bsh[cur][bn * LDS_STR + hi * 8];
            h8 bhh = *(const h8*)&Bsh[cur][bn * LDS_STR + hi * 8 + 16];
            bf[cc] = __builtin_shufflevector(blo, bhh, 0, 1, 2, 3, 4, 5, 6, 7,
                                             8, 9, 10, 11, 12, 13, 14, 15);
        }
        // ---- 8 WMMAs per wave per k-step ----
        #pragma unroll
        for (int rr = 0; rr < 4; ++rr)
            #pragma unroll
            for (int cc = 0; cc < 2; ++cc)
                acc[rr][cc] = __builtin_amdgcn_wmma_f32_16x16x32_f16(
                    false, af[rr], false, bf[cc], (short)0, acc[rr][cc],
                    false, false);

        // Convert+store next tile into the other buffer (waits on global loads
        // only here, after the WMMAs were issued).
        if (st + 1 < nsteps) commit(cur ^ 1);
        __syncthreads();
    }

    // ---- Epilogue. D layout: lane l -> n = l&15, m = (l>>4)*8 + vgpr ----
    #pragma unroll
    for (int rr = 0; rr < 4; ++rr) {
        #pragma unroll
        for (int cc = 0; cc < 2; ++cc) {
            int ncol = tileN + 16 * (wcol0 + cc) + l15;
            if (!FULLN && ncol >= N) continue;
            float bv2 = bias ? bias[ncol] : 0.f;
            #pragma unroll
            for (int v = 0; v < 8; ++v) {
                int m = tileM + 16 * (wrow0 + rr) + hi * 8 + v;
                float val = acc[rr][cc][v] + bv2;
                if (gelu) val = 0.5f * val * (1.f + erff(val * 0.70710678118654752f));
                if (resid) val += resid[(size_t)m * N + ncol];
                C[(size_t)m * N + ncol] = val;
            }
        }
    }
}

// ---------------------------------------------------------------------------
// LayerNorm: one wave32 per token, float4 vectorized
// ---------------------------------------------------------------------------
__global__ __launch_bounds__(256) void ln_kernel(
    const float* __restrict__ x, const float* __restrict__ w,
    const float* __restrict__ b, float* __restrict__ y, int T, int C)
{
    int wid  = threadIdx.x >> 5;
    int lane = threadIdx.x & 31;
    int token = blockIdx.x * 8 + wid;
    if (token >= T) return;
    const float* xr = x + (size_t)token * C;
    float s = 0.f, ss = 0.f;
    for (int i = lane * 4; i < C; i += 128) {
        f4 v = *(const f4*)(xr + i);
        s  += v[0] + v[1] + v[2] + v[3];
        ss += v[0]*v[0] + v[1]*v[1] + v[2]*v[2] + v[3]*v[3];
    }
    #pragma unroll
    for (int o = 16; o > 0; o >>= 1) {
        s  += __shfl_xor(s,  o, 32);
        ss += __shfl_xor(ss, o, 32);
    }
    float mu  = s / C;
    float var = ss / C - mu * mu;
    float inv = rsqrtf(var + 1e-5f);
    float* yr = y + (size_t)token * C;
    for (int i = lane * 4; i < C; i += 128) {
        f4 v = *(const f4*)(xr + i);
        f4 wv = *(const f4*)(w + i);
        f4 bb = *(const f4*)(b + i);
        f4 r;
        #pragma unroll
        for (int u = 0; u < 4; ++u) r[u] = (v[u] - mu) * inv * wv[u] + bb[u];
        *(f4*)(yr + i) = r;
    }
}

// ---------------------------------------------------------------------------
// Windowed attention: one block (64 threads) per (batch, window, head).
// Roll / window-partition / inverse-roll folded into indexing.
// Relative-position index and Swin shift mask computed inline.
// head_dim == 32 for both stages.
// ---------------------------------------------------------------------------
__global__ __launch_bounds__(64) void attn_kernel(
    const float* __restrict__ qkv,  // (B*L, 3C)
    const float* __restrict__ rpb,  // (225, nh)
    float* __restrict__ out,        // (B*L, C)
    int Hdim, int C, int nh, int shift)
{
    const int nWx = Hdim >> 3;
    const int nW  = nWx * nWx;
    int bx   = blockIdx.x;
    int head = bx % nh;
    int win  = (bx / nh) % nW;
    int b    = bx / (nh * nW);
    int wy = win / nWx, wx = win % nWx;

    __shared__ float kS[64][36];
    __shared__ float vS[64][36];
    __shared__ float sS[64][65];

    int i  = threadIdx.x;
    int iy = i >> 3, ix = i & 7;
    // source token in un-rolled coordinates (roll by -shift)
    int sy = (wy * 8 + iy + shift) % Hdim;
    int sx = (wx * 8 + ix + shift) % Hdim;
    size_t tok = (size_t)(b * Hdim + sy) * Hdim + sx;
    const float* qp = qkv + tok * (size_t)(3 * C) + head * 32;

    const float scale = 0.17677669529663687f; // 32^-0.5
    float q[32];
    #pragma unroll
    for (int d4 = 0; d4 < 8; ++d4) {
        f4 v = *(const f4*)(qp + d4 * 4);
        #pragma unroll
        for (int u = 0; u < 4; ++u) q[d4 * 4 + u] = v[u] * scale;
    }
    #pragma unroll
    for (int d4 = 0; d4 < 8; ++d4)
        *(f4*)&kS[i][d4 * 4] = *(const f4*)(qp + C + d4 * 4);
    #pragma unroll
    for (int d4 = 0; d4 < 8; ++d4)
        *(f4*)&vS[i][d4 * 4] = *(const f4*)(qp + 2 * C + d4 * 4);
    __syncthreads();

    int reg_i = 0;
    if (shift) {
        int gy = wy * 8 + iy, gx = wx * 8 + ix;
        int rh = (gy < Hdim - 8) ? 0 : (gy < Hdim - 4 ? 1 : 2);
        int rw = (gx < Hdim - 8) ? 0 : (gx < Hdim - 4 ? 1 : 2);
        reg_i = rh * 3 + rw;
    }
    for (int j = 0; j < 64; ++j) {
        float acc = 0.f;
        #pragma unroll
        for (int d = 0; d < 32; ++d) acc += q[d] * kS[j][d];
        int jy = j >> 3, jx = j & 7;
        acc += rpb[((iy - jy + 7) * 15 + (ix - jx + 7)) * nh + head];
        if (shift) {
            int gy = wy * 8 + jy, gx = wx * 8 + jx;
            int rh = (gy < Hdim - 8) ? 0 : (gy < Hdim - 4 ? 1 : 2);
            int rw = (gx < Hdim - 8) ? 0 : (gx < Hdim - 4 ? 1 : 2);
            if (rh * 3 + rw != reg_i) acc -= 100.f;
        }
        sS[i][j] = acc;
    }
    float mx = -1e30f;
    for (int j = 0; j < 64; ++j) mx = fmaxf(mx, sS[i][j]);
    float sum = 0.f;
    for (int j = 0; j < 64; ++j) { float e = __expf(sS[i][j] - mx); sS[i][j] = e; sum += e; }
    float rs = 1.f / sum;

    float o[32];
    #pragma unroll
    for (int d = 0; d < 32; ++d) o[d] = 0.f;
    for (int j = 0; j < 64; ++j) {
        float w = sS[i][j] * rs;
        #pragma unroll
        for (int d = 0; d < 32; ++d) o[d] += w * vS[j][d];
    }
    float* op = out + tok * (size_t)C + head * 32;
    #pragma unroll
    for (int d4 = 0; d4 < 8; ++d4) {
        f4 v;
        #pragma unroll
        for (int u = 0; u < 4; ++u) v[u] = o[d4 * 4 + u];
        *(f4*)(op + d4 * 4) = v;
    }
}

// ---------------------------------------------------------------------------
// PatchReverse pixel-shuffle: y(B,H,W,4*od) -> x(B,2H,2W,od),
// out[b,2h+i,2w+j,c] = y[b,h,w,c*4+i*2+j]
// ---------------------------------------------------------------------------
__global__ void shuffle_kernel(const float* __restrict__ y, float* __restrict__ x,
                               int Bn, int Hh, int Ww, int od)
{
    size_t idx = (size_t)blockIdx.x * blockDim.x + threadIdx.x;
    size_t total = (size_t)Bn * Hh * Ww * 4 * od;
    if (idx >= total) return;
    int c = (int)(idx % od); size_t r = idx / od;
    int xx = (int)(r % (2 * Ww)); r /= (size_t)(2 * Ww);
    int yy = (int)(r % (2 * Hh)); int b = (int)(r / (2 * Hh));
    int h = yy >> 1, ii = yy & 1, w = xx >> 1, jj = xx & 1;
    x[idx] = y[((size_t)(b * Hh + h) * Ww + w) * (size_t)(4 * od) + c * 4 + ii * 2 + jj];
}

// x(B,65536,3) -> out(B,3,256,256)
__global__ void to_nchw(const float* __restrict__ x, float* __restrict__ out)
{
    size_t idx = (size_t)blockIdx.x * blockDim.x + threadIdx.x;
    size_t total = (size_t)8 * 3 * 65536;
    if (idx >= total) return;
    int p = (int)(idx % 65536); size_t r = idx / 65536;
    int cch = (int)(r % 3); int b = (int)(r / 3);
    out[idx] = x[((size_t)b * 65536 + p) * 3 + cch];
}

// ---------------------------------------------------------------------------
static void launch_gemm(const float* A, const float* W, const float* bias,
                        const float* resid, float* C, int M, int N, int K,
                        int gelu, hipStream_t s)
{
    dim3 grid((N + BN - 1) / BN, M / BM);
    if (N % BN == 0)
        gemm_wmma<true><<<grid, 256, 0, s>>>(A, W, bias, resid, C, M, N, K, gelu);
    else
        gemm_wmma<false><<<grid, 256, 0, s>>>(A, W, bias, resid, C, M, N, K, gelu);
}

extern "C" void kernel_launch(void* const* d_in, const int* in_sizes, int n_in,
                              void* d_out, int out_size, void* d_ws, size_t ws_size,
                              hipStream_t stream)
{
    (void)in_sizes; (void)n_in; (void)out_size; (void)ws_size;
    auto P = [&](int i) { return (const float*)d_in[i]; };

    // Workspace layout (floats). Peak: layer1 MLP hidden (8*16384*512 = 64M f).
    float* ws = (float*)d_ws;
    const size_t RX = 16777216ull;           // 64 MB per activation region
    float* X   = ws;                         // current activations
    float* X2  = ws + RX;                    // residual ping-pong
    float* LNb = ws + 2 * RX;                // layernorm output
    float* O   = ws + 3 * RX;                // attention / pr-gemm output
    float* QH  = ws + 4 * RX;                // qkv / mlp-hidden (67108864 f)

    // head: (B*4096, 32) @ (256,32)^T + b -> X
    launch_gemm((const float*)d_in[0], P(1), P(2), nullptr, X,
                32768, 256, 32, 0, stream);

    const int Hs[2]       = {64, 128};
    const int Cs[2]       = {256, 128};
    const int nhs[2]      = {8, 4};
    const int blk_base[2][2] = {{3, 16}, {32, 45}};
    const int up_base[2]  = {29, 58};
    const int out_dims[2] = {128, 3};

    for (int li = 0; li < 2; ++li) {
        int H = Hs[li], C = Cs[li], nh = nhs[li];
        int T  = 8 * H * H;
        int nW = (H / 8) * (H / 8);
        for (int bj = 0; bj < 2; ++bj) {
            int base  = blk_base[li][bj];
            int shift = bj ? 4 : 0;
            // block order: norm1_w/b, qkv_w/b, rpb, proj_w/b, norm2_w/b, fc1_w/b, fc2_w/b
            ln_kernel<<<(T + 7) / 8, 256, 0, stream>>>(X, P(base + 0), P(base + 1), LNb, T, C);
            launch_gemm(LNb, P(base + 2), P(base + 3), nullptr, QH, T, 3 * C, C, 0, stream);
            attn_kernel<<<8 * nW * nh, 64, 0, stream>>>(QH, P(base + 4), O, H, C, nh, shift);
            launch_gemm(O, P(base + 5), P(base + 6), /*resid=*/X, X2, T, C, C, 0, stream);
            ln_kernel<<<(T + 7) / 8, 256, 0, stream>>>(X2, P(base + 7), P(base + 8), LNb, T, C);
            launch_gemm(LNb, P(base + 9), P(base + 10), nullptr, QH, T, 4 * C, C, /*gelu=*/1, stream);
            launch_gemm(QH, P(base + 11), P(base + 12), /*resid=*/X2, X, T, C, 4 * C, 0, stream);
        }
        // PatchReverseMerging: LN -> Linear(dim, 4*od) -> PixelShuffle(2)
        int ub = up_base[li], od = out_dims[li];
        ln_kernel<<<(T + 7) / 8, 256, 0, stream>>>(X, P(ub + 0), P(ub + 1), LNb, T, C);
        launch_gemm(LNb, P(ub + 2), nullptr, nullptr, O, T, 4 * od, C, 0, stream);
        size_t tot = (size_t)4 * T * od;
        shuffle_kernel<<<(unsigned)((tot + 255) / 256), 256, 0, stream>>>(O, X, 8, H, H, od);
    }
    size_t tot = (size_t)8 * 3 * 65536;
    to_nchw<<<(unsigned)((tot + 255) / 256), 256, 0, stream>>>(X, (float*)d_out);
}
